// DiagonalSSM_47708496724572
// MI455X (gfx1250) — matrix operand order repaired and checked by reading, not compiled, not run on hardware
//
#include <hip/hip_runtime.h>
#include <hip/hip_bf16.h>

typedef __attribute__((ext_vector_type(16))) __bf16 v16bf;
typedef __attribute__((ext_vector_type(8)))  __bf16 v8bf;
typedef __attribute__((ext_vector_type(8)))  float  v8f;
typedef __attribute__((ext_vector_type(4)))  int    v4i;

#if defined(__has_builtin)
#if __has_builtin(__builtin_amdgcn_global_load_async_to_lds_b128)
#define HAVE_ASYNC_LDS 1
#endif
#endif

#ifdef HAVE_ASYNC_LDS
typedef v4i __attribute__((address_space(1))) * as1_v4i_p;   // global int4*
typedef v4i __attribute__((address_space(3))) * as3_v4i_p;   // LDS int4*
#endif

#define T_    8
#define V_    10000
#define E_    100000
#define CIN_  128
#define H_    256
#define DS_   16
#define COUT_ 64
#define XSTRIDE 256   // all activation buffers padded to 256 channels

// ---------------------------------------------------------------- utilities
__global__ void fill_f32(float* p, float v, long n) {
    long i = (long)blockIdx.x * blockDim.x + threadIdx.x;
    if (i < n) p[i] = v;
}

// one-time: W[K,N] f32 -> WT[N,K] bf16
__global__ void transpose_to_bf16(const float* __restrict__ W, __bf16* __restrict__ WT,
                                  int K, int N) {
    long i = (long)blockIdx.x * blockDim.x + threadIdx.x;
    if (i >= (long)K * N) return;
    int k = (int)(i / N), n = (int)(i % N);           // coalesced read over n
    WT[(long)n * K + k] = (__bf16)W[(long)k * N + n];
}

// depthwise conv1d over time, kernel 3, zero pad 1.  out stride = XSTRIDE
__global__ void token_mix(const float* __restrict__ xs,
                          const float* __restrict__ w,
                          const float* __restrict__ b,
                          float* __restrict__ X) {
    long i = (long)blockIdx.x * blockDim.x + threadIdx.x;
    if (i >= (long)T_ * V_ * CIN_) return;
    int c = (int)(i % CIN_);
    long tv = i / CIN_;
    int v = (int)(tv % V_);
    int t = (int)(tv / V_);
    float acc = b[c];
    if (t - 1 >= 0) acc += xs[((long)(t - 1) * V_ + v) * CIN_ + c] * w[c * 3 + 0];
    acc += xs[((long)t * V_ + v) * CIN_ + c] * w[c * 3 + 1];
    if (t + 1 < T_) acc += xs[((long)(t + 1) * V_ + v) * CIN_ + c] * w[c * 3 + 2];
    X[((long)t * V_ + v) * XSTRIDE + c] = acc;
}

// ---------------------------------------------------------------- SAGE aggregation
__global__ void edge_deg(const long long* __restrict__ ei, float* __restrict__ deg, int t) {
    int e = blockIdx.x * blockDim.x + threadIdx.x;
    if (e >= E_) return;
    int dst = (int)ei[((long)t * 2 + 1) * E_ + e];
    atomicAdd(&deg[dst], 1.0f);
}

// scatter-add x[src] into agg[dst]; one thread handles 4 channels of one edge
__global__ void edge_scatter(const long long* __restrict__ ei,
                             const float* __restrict__ X,
                             float* __restrict__ agg, int t, int C) {
    long i = (long)blockIdx.x * blockDim.x + threadIdx.x;
    int nq = C >> 2;
    if (i >= (long)E_ * nq) return;
    int e  = (int)(i / nq);
    int cq = (int)(i % nq) << 2;
    int src = (int)ei[((long)t * 2 + 0) * E_ + e];
    int dst = (int)ei[((long)t * 2 + 1) * E_ + e];
    float4 xv = *(const float4*)&X[((long)t * V_ + src) * XSTRIDE + cq];
    float* ap = &agg[(long)dst * XSTRIDE + cq];
    atomicAdd(ap + 0, xv.x);
    atomicAdd(ap + 1, xv.y);
    atomicAdd(ap + 2, xv.z);
    atomicAdd(ap + 3, xv.w);
}

__global__ void mean_div(float* __restrict__ agg, const float* __restrict__ deg, int C) {
    long i = (long)blockIdx.x * blockDim.x + threadIdx.x;
    if (i >= (long)V_ * C) return;
    int v = (int)(i / C), c = (int)(i % C);
    agg[(long)v * XSTRIDE + c] /= fmaxf(deg[v], 1.0f);
}

// ---------------------------------------------------------------- SSM pieces
__global__ void compute_lam(const float* __restrict__ a_log, float* __restrict__ lam) {
    int i = blockIdx.x * blockDim.x + threadIdx.x;
    if (i < H_ * DS_) lam[i] = __expf(-__expf(a_log[i]));
}

// state[v,h,:] = lam[h,:]*state + h_t[v,h]*B[h,:]; emit relu(state) as bf16 row of A
__global__ void ssm_step(float* __restrict__ state,
                         const float* __restrict__ Hbuf,
                         const float* __restrict__ lam,
                         const float* __restrict__ Bm,
                         __bf16* __restrict__ Abuf, int t) {
    long i = (long)blockIdx.x * blockDim.x + threadIdx.x;
    if (i >= (long)V_ * H_) return;
    int v  = (int)(i / H_);
    int hh = (int)(i % H_);
    float x = Hbuf[((long)t * V_ + v) * XSTRIDE + hh];
    float* st = &state[((long)v * H_ + hh) * DS_];
    const float* lm = &lam[hh * DS_];
    const float* bp = &Bm[hh * DS_];
    __bf16 outv[DS_];
#pragma unroll
    for (int d = 0; d < DS_; ++d) {
        float s = lm[d] * st[d] + x * bp[d];
        st[d] = s;
        outv[d] = (__bf16)fmaxf(s, 0.0f);
    }
    uint4* dp = (uint4*)&Abuf[(long)v * (H_ * DS_) + (long)hh * DS_];
    dp[0] = ((const uint4*)outv)[0];
    dp[1] = ((const uint4*)outv)[1];
}

// ---------------------------------------------------------------- WMMA GEMM
// C[M,N] (ldc) = A[M,K] (lda) @ WT^T  where WT is [N,K] bf16 (pre-transposed weights)
// flags: 1 = accumulate into existing C, 2 = add bias[col], 4 = add Dadd[row,col]
// Block: 256 thr = 8 waves; tile 128x64; each wave -> 32x32 (4 x v_wmma 16x16x32 bf16)
// Double-buffered LDS pipeline: stage tile k+1 (async where possible) while WMMA on tile k.
template <bool ABF16>
__global__ __launch_bounds__(256)
void gemm_wmma(const void* __restrict__ Av, int lda,
               const __bf16* __restrict__ WT,
               const float* __restrict__ bias,
               const float* __restrict__ Dadd, int ldd,
               float* __restrict__ C, int ldc,
               int M, int N, int K, int flags) {
    __shared__ __align__(32) __bf16 As[2][128][40];   // 32 K + 8 pad  (80 B rows)
    __shared__ __align__(32) __bf16 Bs[2][64][48];    // [n][k], 32 K + 16 pad (96 B rows)

    const int tid  = threadIdx.x;
    const int lane = tid & 31;
    const int wave = tid >> 5;
    const int wm   = wave & 3;       // M subtile (x32)
    const int wn   = wave >> 2;      // N subtile (x32)
    const int m0   = blockIdx.x * 128;
    const int n0   = blockIdx.y * 64;
    const int l16  = lane & 15;
    const int hi   = lane >> 4;

    v8f acc[2][2] = {};

    const int arow = tid >> 1;               // 0..127
    const int akk  = (tid & 1) * 16;         // 0 / 16
    const int bn   = tid & 63;               // 0..63  (N column)
    const int bks  = (tid >> 6) * 8;         // 0,8,16,24 (K segment)
    const long gm  = m0 + arow;
    const bool arow_ok = (gm < M);

    auto stage = [&](int buf, int k0) {
        // --- A tile: 16 elements/thread ---
        if (ABF16) {
            const __bf16* ap = (const __bf16*)Av + gm * (long)lda + k0 + akk;
            if (arow_ok) {
#ifdef HAVE_ASYNC_LDS
                __builtin_amdgcn_global_load_async_to_lds_b128(
                    (as1_v4i_p)(void*)ap,       (as3_v4i_p)(void*)&As[buf][arow][akk],     0, 0);
                __builtin_amdgcn_global_load_async_to_lds_b128(
                    (as1_v4i_p)(void*)(ap + 8), (as3_v4i_p)(void*)&As[buf][arow][akk + 8], 0, 0);
#else
                const uint4* a4 = (const uint4*)ap;
                *(uint4*)&As[buf][arow][akk]     = a4[0];
                *(uint4*)&As[buf][arow][akk + 8] = a4[1];
#endif
            } else {
                uint4 z; z.x = z.y = z.z = z.w = 0u;
                *(uint4*)&As[buf][arow][akk]     = z;
                *(uint4*)&As[buf][arow][akk + 8] = z;
            }
        } else {
            if (arow_ok) {
                const float4* ap = (const float4*)((const float*)Av + gm * (long)lda + k0 + akk);
                float4 f0 = ap[0], f1 = ap[1], f2 = ap[2], f3 = ap[3];
                v8bf t0, t1;
                t0[0] = (__bf16)f0.x; t0[1] = (__bf16)f0.y; t0[2] = (__bf16)f0.z; t0[3] = (__bf16)f0.w;
                t0[4] = (__bf16)f1.x; t0[5] = (__bf16)f1.y; t0[6] = (__bf16)f1.z; t0[7] = (__bf16)f1.w;
                t1[0] = (__bf16)f2.x; t1[1] = (__bf16)f2.y; t1[2] = (__bf16)f2.z; t1[3] = (__bf16)f2.w;
                t1[4] = (__bf16)f3.x; t1[5] = (__bf16)f3.y; t1[6] = (__bf16)f3.z; t1[7] = (__bf16)f3.w;
                *(v8bf*)&As[buf][arow][akk]     = t0;
                *(v8bf*)&As[buf][arow][akk + 8] = t1;
                // prefetch the slab after next (global_prefetch_b8)
                if (k0 + 32 < K)
                    __builtin_prefetch((const float*)Av + gm * (long)lda + k0 + 32 + akk, 0, 1);
            } else {
                uint4 z; z.x = z.y = z.z = z.w = 0u;
                *(uint4*)&As[buf][arow][akk]     = z;
                *(uint4*)&As[buf][arow][akk + 8] = z;
            }
        }
        // --- B tile: pre-transposed bf16, 16 B/thread ---
        const __bf16* wp = WT + (long)(n0 + bn) * K + k0 + bks;
#ifdef HAVE_ASYNC_LDS
        __builtin_amdgcn_global_load_async_to_lds_b128(
            (as1_v4i_p)(void*)wp, (as3_v4i_p)(void*)&Bs[buf][bn][bks], 0, 0);
#else
        *(uint4*)&Bs[buf][bn][bks] = *(const uint4*)wp;
#endif
    };

    stage(0, 0);
    int p = 0;
    for (int k0 = 0; k0 < K; k0 += 32) {
#ifdef HAVE_ASYNC_LDS
        asm volatile("s_wait_asynccnt 0x0" ::: "memory");
#endif
        __syncthreads();
        if (k0 + 32 < K) stage(p ^ 1, k0 + 32);   // overlap next-tile staging with WMMA

        // --- WMMA fragments: element s of lane half hi holds K = 16*hi + s, same order in A and B ---
        v16bf afrag[2], bfrag[2];
#pragma unroll
        for (int mi = 0; mi < 2; ++mi) {
            int r = wm * 32 + mi * 16 + l16;                      // matrix row for this lane
            v8bf lo = *(const v8bf*)&As[p][r][hi * 16];           // K = 16*hi .. 16*hi+7
            v8bf hv = *(const v8bf*)&As[p][r][hi * 16 + 8];       // K = 16*hi+8 .. 16*hi+15
#pragma unroll
            for (int j = 0; j < 8; ++j) { afrag[mi][j] = lo[j]; afrag[mi][j + 8] = hv[j]; }
        }
#pragma unroll
        for (int ni = 0; ni < 2; ++ni) {
            int c = wn * 32 + ni * 16 + l16;                      // matrix col for this lane
            bfrag[ni] = *(const v16bf*)&Bs[p][c][hi * 16];        // 16 contiguous K
        }
#pragma unroll
        for (int mi = 0; mi < 2; ++mi)
#pragma unroll
            for (int ni = 0; ni < 2; ++ni)
                acc[mi][ni] = __builtin_amdgcn_wmma_f32_16x16x32_bf16(
                    false, afrag[mi], false, bfrag[ni],
                    (short)0, acc[mi][ni], false, false);
        p ^= 1;
    }

    // --- epilogue: C frag layout: VGPR r -> row r + 8*hi ; col = lane%16 ---
    const bool full = (m0 + 128 <= M);
#pragma unroll
    for (int mi = 0; mi < 2; ++mi) {
#pragma unroll
        for (int ni = 0; ni < 2; ++ni) {
            int col = n0 + wn * 32 + ni * 16 + l16;
            int rbase = m0 + wm * 32 + mi * 16 + hi * 8;
            if (full) {
#pragma unroll
                for (int r = 0; r < 8; ++r) {
                    int row = rbase + r;
                    float v = acc[mi][ni][r];
                    if (flags & 2) v += bias[col];
                    if (flags & 4) v += Dadd[(long)row * ldd + col];
                    long ci = (long)row * ldc + col;
                    if (flags & 1) v += C[ci];
                    C[ci] = v;
                }
            } else {
#pragma unroll
                for (int r = 0; r < 8; ++r) {
                    int row = rbase + r;
                    if (row < M) {
                        float v = acc[mi][ni][r];
                        if (flags & 2) v += bias[col];
                        if (flags & 4) v += Dadd[(long)row * ldd + col];
                        long ci = (long)row * ldc + col;
                        if (flags & 1) v += C[ci];
                        C[ci] = v;
                    }
                }
            }
        }
    }
}

// ---------------------------------------------------------------- host side
static inline unsigned cdiv(long n, int b) { return (unsigned)((n + b - 1) / b); }

extern "C" void kernel_launch(void* const* d_in, const int* in_sizes, int n_in,
                              void* d_out, int out_size, void* d_ws, size_t ws_size,
                              hipStream_t stream) {
    (void)in_sizes; (void)n_in; (void)out_size; (void)ws_size;

    const float*     xs    = (const float*)d_in[0];
    const long long* ei    = (const long long*)d_in[1];
    const float*     w_pre = (const float*)d_in[2];
    const float*     b_pre = (const float*)d_in[3];
    const float* w_res[2]   = {(const float*)d_in[4],  (const float*)d_in[13]};
    const float* b_res[2]   = {(const float*)d_in[5],  (const float*)d_in[14]};
    const float* w_self[2]  = {(const float*)d_in[6],  (const float*)d_in[15]};
    const float* w_neigh[2] = {(const float*)d_in[7],  (const float*)d_in[16]};
    const float* b_sage[2]  = {(const float*)d_in[8],  (const float*)d_in[17]};
    const float* a_log[2]   = {(const float*)d_in[9],  (const float*)d_in[18]};
    const float* Bmat[2]    = {(const float*)d_in[10], (const float*)d_in[19]};
    const float* w_mix[2]   = {(const float*)d_in[11], (const float*)d_in[20]};
    const float* b_mix[2]   = {(const float*)d_in[12], (const float*)d_in[21]};
    const float* w_out = (const float*)d_in[22];
    const float* b_out = (const float*)d_in[23];
    float* out = (float*)d_out;

    // workspace carve-up (256B aligned)
    char* ws = (char*)d_ws;
    size_t off = 0;
    auto take = [&](size_t bytes) { char* p = ws + off; off = (off + bytes + 255) & ~(size_t)255; return p; };
    float*  X     = (float*)take((size_t)T_ * V_ * XSTRIDE * 4);   // activations (in/out of layer)
    float*  XSR   = (float*)take((size_t)T_ * V_ * XSTRIDE * 4);   // residual branch
    float*  Hb    = (float*)take((size_t)T_ * V_ * XSTRIDE * 4);   // sage output
    float*  STATE = (float*)take((size_t)V_ * H_ * DS_ * 4);       // SSM state f32
    __bf16* AB    = (__bf16*)take((size_t)V_ * H_ * DS_ * 2);      // relu(state) as bf16 GEMM-A
    float*  AGG   = (float*)take((size_t)V_ * XSTRIDE * 4);        // per-t neighbor mean
    float*  DEG   = (float*)take((size_t)V_ * 4);
    float*  LAM   = (float*)take((size_t)H_ * DS_ * 4);
    // pre-transposed bf16 weights: WT[N][K]
    __bf16* WTres[2], *WTself[2], *WTneigh[2], *WTmix[2];
    for (int L = 0; L < 2; ++L) {
        int Cin = (L == 0) ? CIN_ : H_;
        WTres[L]   = (__bf16*)take((size_t)H_ * Cin * 2);
        WTself[L]  = (__bf16*)take((size_t)H_ * Cin * 2);
        WTneigh[L] = (__bf16*)take((size_t)H_ * Cin * 2);
        WTmix[L]   = (__bf16*)take((size_t)H_ * (H_ * DS_) * 2);
    }
    __bf16* WTout = (__bf16*)take((size_t)COUT_ * H_ * 2);

    const int TB = 256;

    // 0) one-time weight transpose + bf16 conversion
    for (int L = 0; L < 2; ++L) {
        int Cin = (L == 0) ? CIN_ : H_;
        transpose_to_bf16<<<cdiv((long)Cin * H_, TB), TB, 0, stream>>>(w_res[L],   WTres[L],   Cin, H_);
        transpose_to_bf16<<<cdiv((long)Cin * H_, TB), TB, 0, stream>>>(w_self[L],  WTself[L],  Cin, H_);
        transpose_to_bf16<<<cdiv((long)Cin * H_, TB), TB, 0, stream>>>(w_neigh[L], WTneigh[L], Cin, H_);
        transpose_to_bf16<<<cdiv((long)H_ * DS_ * H_, TB), TB, 0, stream>>>(w_mix[L], WTmix[L], H_ * DS_, H_);
    }
    transpose_to_bf16<<<cdiv((long)H_ * COUT_, TB), TB, 0, stream>>>(w_out, WTout, H_, COUT_);

    // 1) token mixer
    token_mix<<<cdiv((long)T_ * V_ * CIN_, TB), TB, 0, stream>>>(xs, w_pre, b_pre, X);

    // 2) two layers
    for (int L = 0; L < 2; ++L) {
        const int Cin = (L == 0) ? CIN_ : H_;
        const long MT = (long)T_ * V_;

        // residual + self projections over all T at once
        gemm_wmma<false><<<dim3(cdiv(MT, 128), H_ / 64), TB, 0, stream>>>(
            X, XSTRIDE, WTres[L], b_res[L], nullptr, 0, XSR, XSTRIDE,
            (int)MT, H_, Cin, /*bias*/2);
        gemm_wmma<false><<<dim3(cdiv(MT, 128), H_ / 64), TB, 0, stream>>>(
            X, XSTRIDE, WTself[L], b_sage[L], nullptr, 0, Hb, XSTRIDE,
            (int)MT, H_, Cin, /*bias*/2);

        // per-t neighbor mean, accumulated into Hb
        for (int t = 0; t < T_; ++t) {
            fill_f32<<<cdiv((long)V_ * XSTRIDE, TB), TB, 0, stream>>>(AGG, 0.0f, (long)V_ * XSTRIDE);
            fill_f32<<<cdiv(V_, TB), TB, 0, stream>>>(DEG, 0.0f, V_);
            edge_deg<<<cdiv(E_, TB), TB, 0, stream>>>(ei, DEG, t);
            edge_scatter<<<cdiv((long)E_ * (Cin / 4), TB), TB, 0, stream>>>(ei, X, AGG, t, Cin);
            mean_div<<<cdiv((long)V_ * Cin, TB), TB, 0, stream>>>(AGG, DEG, Cin);
            gemm_wmma<false><<<dim3(cdiv(V_, 128), H_ / 64), TB, 0, stream>>>(
                AGG, XSTRIDE, WTneigh[L], nullptr, nullptr, 0,
                Hb + (long)t * V_ * XSTRIDE, XSTRIDE, V_, H_, Cin, /*accum*/1);
        }

        // SSM scan: state recurrence + mix GEMM, output overwrites X slice t
        compute_lam<<<cdiv(H_ * DS_, TB), TB, 0, stream>>>(a_log[L], LAM);
        fill_f32<<<cdiv((long)V_ * H_ * DS_, TB), TB, 0, stream>>>(STATE, 0.0f, (long)V_ * H_ * DS_);
        for (int t = 0; t < T_; ++t) {
            ssm_step<<<cdiv((long)V_ * H_, TB), TB, 0, stream>>>(STATE, Hb, LAM, Bmat[L], AB, t);
            gemm_wmma<true><<<dim3(cdiv(V_, 128), H_ / 64), TB, 0, stream>>>(
                AB, H_ * DS_, WTmix[L], b_mix[L],
                XSR + (long)t * V_ * XSTRIDE, XSTRIDE,            // + residual
                X + (long)t * V_ * XSTRIDE, XSTRIDE,              // write layer output
                V_, H_, H_ * DS_, /*bias|Dadd*/ 2 | 4);
        }
    }

    // 3) output projection from last time step
    gemm_wmma<false><<<dim3(cdiv(V_, 128), COUT_ / 64), TB, 0, stream>>>(
        X + (long)(T_ - 1) * V_ * XSTRIDE, XSTRIDE, WTout, b_out,
        nullptr, 0, out, COUT_, V_, COUT_, H_, /*bias*/2);
}
